// Block_65206193488488
// MI455X (gfx1250) — matrix operand order repaired and checked by reading, not compiled, not run on hardware
//
#include <hip/hip_runtime.h>

// ---------------------------------------------------------------------------
// Transformer-XL layer for MI455X (gfx1250, wave32, WMMA + TDM).
// B=4, Q=512, M=512, D=1024, H=16, Kd=64, KL=M+Q=1024, F=4*D=4096.
// fp16 WMMA (f32 accum) for every GEMM; GEMM tiles staged into LDS by the
// Tensor Data Mover (double-buffered, s_wait_tensorcnt); flash-style fused
// attention with exact Transformer-XL rel_shift indexing.
// ---------------------------------------------------------------------------

constexpr int Bb = 4;
constexpr int Qq = 512;
constexpr int Mm = 512;
constexpr int KL = 1024;   // M + Q
constexpr int Dd = 1024;
constexpr int Hh = 16;
constexpr int Kd = 64;
constexpr int HK = 1024;   // H*Kd
constexpr int Ff = 4096;   // 4*D

typedef _Float16 v16h __attribute__((ext_vector_type(16)));
typedef _Float16 v8h  __attribute__((ext_vector_type(8)));
typedef float    v8f  __attribute__((ext_vector_type(8)));
typedef unsigned int u32x4 __attribute__((ext_vector_type(4)));
typedef int          i32x4 __attribute__((ext_vector_type(4)));
typedef int          i32x8 __attribute__((ext_vector_type(8)));

__device__ inline v8f wmma_f16(v16h a, v16h b, v8f c) {
  // (neg_a, A, neg_b, B, c_mod, C, reuse_a, reuse_b)
  return __builtin_amdgcn_wmma_f32_16x16x32_f16(false, a, false, b, (short)0, c,
                                                false, false);
}

// --------------------------- TDM tile loader -------------------------------
// 2-D tile DMA: global (row-major, elem=2B, row stride strideElems) -> LDS at
// ldsOff, tile = tileRows x 32 halfs, with 4-DWORD LDS padding after each
// 16-DWORD (32-half) row => LDS row pitch of 40 halfs. D# per ISA 8.3/8.4.
__device__ inline void tdm_load_tile(unsigned ldsOff, const _Float16* gsrc,
                                     unsigned strideElems, unsigned tileRows) {
  unsigned long long ga = (unsigned long long)(const void*)gsrc;
  u32x4 g0;
  g0[0] = 1u;                                   // count=1, user descriptor
  g0[1] = ldsOff;                               // lds_addr
  g0[2] = (unsigned)(ga & 0xffffffffu);         // global_addr[31:0]
  g0[3] = (unsigned)((ga >> 32) & 0x01ffffffu)  // global_addr[56:32]
          | (2u << 30);                         // type = 2 ("image")
  i32x8 g1;
  g1[0] = (int)((1u << 16)                      // data_size = 1 -> 2 bytes
                | (1u << 20)                    // pad_enable
                | (3u << 22)                    // pad_interval: 16 DWORDs
                | (3u << 25));                  // pad_amount:   4 DWORDs
  g1[1] = (int)((strideElems & 0xffffu) << 16); // tensor_dim0 lo16
  g1[2] = (int)((strideElems >> 16) | ((tileRows & 0xffffu) << 16)); // d0 hi | d1 lo
  g1[3] = (int)((tileRows >> 16) | (32u << 16)); // d1 hi | tile_dim0 = 32
  g1[4] = (int)(tileRows & 0xffffu);             // tile_dim1 (tile_dim2 = 0)
  g1[5] = (int)strideElems;                      // tensor_dim0_stride lo32
  g1[6] = 0;
  g1[7] = 0;
  i32x4 z4 = {0, 0, 0, 0};
#if defined(__clang_major__) && __clang_major__ >= 23
  i32x8 z8 = {0, 0, 0, 0, 0, 0, 0, 0};
  __builtin_amdgcn_tensor_load_to_lds(g0, g1, z4, z4, z8, 0);
#else
  __builtin_amdgcn_tensor_load_to_lds(g0, g1, z4, z4, 0);
#endif
}

// A-operand fragment (16x32 f16, M x K). ISA 7.12.2: lanes 0-15 hold K{0-7,16-23},
// lanes 16-31 hold K{8-15,24-31}; lane%16 = M row.  src row-major, K fast.
__device__ inline v16h frag_ld(const _Float16* __restrict__ p0, int ld) {
  int lane = threadIdx.x & 31;
  const _Float16* p = p0 + (size_t)(lane & 15) * ld + ((lane >> 4) << 3);
  v8h lo = *(const v8h*)p;
  v8h hi = *(const v8h*)(p + 16);
  v16h r;
#pragma unroll
  for (int e = 0; e < 8; e++) { r[e] = lo[e]; r[e + 8] = hi[e]; }
  return r;
}

// B-operand fragment (32x16 f16, K x N): lanes 0-15 hold K{0-7,16-23}, lanes 16-31
// hold K{8-15,24-31}; lane%16 = N column. src row-major [N, ld], K fast.
__device__ inline v16h frag_ldb(const _Float16* __restrict__ p0, int ld) {
  int lane = threadIdx.x & 31;
  const _Float16* p = p0 + (size_t)(lane & 15) * ld + ((lane >> 4) << 3);
  v8h lo = *(const v8h*)p;
  v8h hi = *(const v8h*)(p + 16);
  v16h r;
#pragma unroll
  for (int e = 0; e < 8; e++) { r[e] = lo[e]; r[e + 8] = hi[e]; }
  return r;
}

__device__ inline float rmax16(float v) {
#pragma unroll
  for (int o = 1; o < 16; o <<= 1) v = fmaxf(v, __shfl_xor(v, o, 32));
  return v;
}
__device__ inline float rsum16(float v) {
#pragma unroll
  for (int o = 1; o < 16; o <<= 1) v += __shfl_xor(v, o, 32);
  return v;
}

// ------------------------------ prep kernels -------------------------------

__global__ void pe_kernel(_Float16* __restrict__ peh) {
  long idx = (long)blockIdx.x * blockDim.x + threadIdx.x;
  if (idx >= (long)KL * Dd) return;
  int p = (int)(idx / Dd), d = (int)(idx % Dd);
  float pos = (float)(KL - 1 - p);
  int j = (d < Dd / 2) ? d : d - Dd / 2;
  float inv_freq = expf(-(2.0f * (float)j / (float)Dd) * 9.210340371976184f);
  float ang = pos * inv_freq;
  peh[idx] = (_Float16)((d < Dd / 2) ? sinf(ang) : cosf(ang));
}

__global__ void conv_f16(const float* __restrict__ src, _Float16* __restrict__ dst, long n) {
  long i = (long)blockIdx.x * blockDim.x + threadIdx.x;
  if (i < n) dst[i] = (_Float16)src[i];
}

__global__ void pack_kv(const float* __restrict__ mem, const float* __restrict__ x,
                        _Float16* __restrict__ kvh) {
  long idx = (long)blockIdx.x * blockDim.x + threadIdx.x;
  if (idx >= (long)Bb * KL * Dd) return;
  int d = (int)(idx % Dd);
  int r = (int)((idx / Dd) % KL);
  int b = (int)(idx / ((long)KL * Dd));
  float v = (r < Mm) ? mem[((size_t)b * Mm + r) * Dd + d]
                     : x[((size_t)b * Qq + (r - Mm)) * Dd + d];
  kvh[idx] = (_Float16)v;
}

// dst[c*rows + r] = src[r*cols + c]  (fp32 -> fp16 transpose)
__global__ void tw_kernel(const float* __restrict__ src, _Float16* __restrict__ dst,
                          int rows, int cols) {
  long i = (long)blockIdx.x * blockDim.x + threadIdx.x;
  if (i >= (long)rows * cols) return;
  int c = (int)(i / rows), r = (int)(i % rows);
  dst[i] = (_Float16)src[(size_t)r * cols + c];
}

// ------------------------------- GEMM core ---------------------------------
// C[m,n] = sum_d A[m,d] * Bt[n,d]; block tile 128x64, 8 waves of 32x32, BK=32.
// A/B tiles DMA'd into LDS by the TDM, double-buffered on TENSORcnt.
// EPI: 0 bdraw f16; 1 qc/qp (+biases, head-major); 2 head-major f16;
//      3 f16 bias+relu; 4 f32 (+bias)(+resid); 5 head-major transposed f16.

template <int EPI>
__global__ __launch_bounds__(256) void gemm_wmma(
    const _Float16* __restrict__ A, const _Float16* __restrict__ Bt, int Kdim,
    long aBatch, long bBatch, int bMod, long cBatch,
    _Float16* __restrict__ outH, _Float16* __restrict__ outH2,
    float* __restrict__ outF, const float* __restrict__ bias1,
    const float* __restrict__ bias2, const float* __restrict__ resid,
    int R, int N) {
  __shared__ __align__(16) _Float16 As[2][128 * 40];
  __shared__ __align__(16) _Float16 Bs[2][64 * 40];
  int z = blockIdx.z;
  const _Float16* Ab = A + (size_t)z * aBatch;
  const _Float16* Bw = Bt + (size_t)(z % bMod) * bBatch;
  int m0 = blockIdx.y * 128, n0 = blockIdx.x * 64;
  int t = threadIdx.x;
  int wave = t >> 5, wm = wave & 3, wn = wave >> 2;
  v8f acc[2][2] = {};

  unsigned offA0 = (unsigned)(size_t)(void*)&As[0][0];
  unsigned offA1 = (unsigned)(size_t)(void*)&As[1][0];
  unsigned offB0 = (unsigned)(size_t)(void*)&Bs[0][0];
  unsigned offB1 = (unsigned)(size_t)(void*)&Bs[1][0];

  int nk = Kdim >> 5;
  if (t < 32) {  // wave 0 drives the TDM pipeline (uniform branch)
    tdm_load_tile(offA0, Ab + (size_t)m0 * Kdim, (unsigned)Kdim, 128u);
    tdm_load_tile(offB0, Bw + (size_t)n0 * Kdim, (unsigned)Kdim, 64u);
  }
  for (int ik = 0; ik < nk; ik++) {
    int cur = ik & 1;
    if (t < 32) {
      if (ik + 1 < nk) {  // prefetch next tile into the other buffer
        unsigned oa = (cur ? offA0 : offA1), ob = (cur ? offB0 : offB1);
        tdm_load_tile(oa, Ab + (size_t)m0 * Kdim + (ik + 1) * 32,
                      (unsigned)Kdim, 128u);
        tdm_load_tile(ob, Bw + (size_t)n0 * Kdim + (ik + 1) * 32,
                      (unsigned)Kdim, 64u);
        __builtin_amdgcn_s_wait_tensorcnt(2);  // current tile's 2 DMAs retired
      } else {
        __builtin_amdgcn_s_wait_tensorcnt(0);
      }
    }
    __syncthreads();  // current tile visible to all waves
    v16h a0 = frag_ld(&As[cur][(wm * 32) * 40], 40);
    v16h a1 = frag_ld(&As[cur][(wm * 32 + 16) * 40], 40);
    v16h b0 = frag_ldb(&Bs[cur][(wn * 32) * 40], 40);
    v16h b1 = frag_ldb(&Bs[cur][(wn * 32 + 16) * 40], 40);
    acc[0][0] = wmma_f16(a0, b0, acc[0][0]);
    acc[0][1] = wmma_f16(a0, b1, acc[0][1]);
    acc[1][0] = wmma_f16(a1, b0, acc[1][0]);
    acc[1][1] = wmma_f16(a1, b1, acc[1][1]);
    __syncthreads();  // reads done before this buffer is re-targeted
  }

  int lane = t & 31, lh = lane >> 4, lc = lane & 15;
#pragma unroll
  for (int fm = 0; fm < 2; fm++) {
#pragma unroll
    for (int fn = 0; fn < 2; fn++) {
#pragma unroll
      for (int g = 0; g < 8; g++) {
        int m = m0 + wm * 32 + fm * 16 + g + lh * 8;
        int n = n0 + wn * 32 + fn * 16 + lc;
        float v = acc[fm][fn][g];
        if (EPI == 0) {
          outH[(size_t)z * cBatch + (size_t)m * N + n] = (_Float16)v;
        } else if (EPI == 1) {
          int bb = m / R, i = m % R, h = n >> 6, kd = n & 63;
          size_t o = (((size_t)bb * Hh + h) * R + i) * Kd + kd;
          outH[o]  = (_Float16)(v + bias1[n]);
          outH2[o] = (_Float16)(v + bias2[n]);
        } else if (EPI == 2) {
          int bb = m / R, i = m % R, h = n >> 6, kd = n & 63;
          outH[(((size_t)bb * Hh + h) * R + i) * Kd + kd] = (_Float16)v;
        } else if (EPI == 5) {
          int bb = m / R, i = m % R, h = n >> 6, kd = n & 63;
          outH[(((size_t)bb * Hh + h) * Kd + kd) * (size_t)R + i] = (_Float16)v;
        } else if (EPI == 3) {
          float y = v + bias1[n];
          outH[(size_t)m * N + n] = (_Float16)(y > 0.f ? y : 0.f);
        } else if (EPI == 4) {
          float y = v;
          if (bias1) y += bias1[n];
          if (resid) y += resid[(size_t)m * N + n];
          outF[(size_t)m * N + n] = y;
        }
      }
    }
  }
}

// --------------------------- fused attention -------------------------------
// One workgroup = (b, h, 64 q-rows); 4 waves, 16 q-rows each. Streams KL in
// tiles of 32 with online softmax; BD read through exact rel_shift indexing.

__global__ __launch_bounds__(128) void attn_kernel(
    const _Float16* __restrict__ qc, const _Float16* __restrict__ kh,
    const _Float16* __restrict__ vth, const _Float16* __restrict__ bdraw,
    const unsigned char* __restrict__ maskp, _Float16* __restrict__ ctxh) {
  __shared__ __align__(16) _Float16 Pl[4][16][40];
  int b = blockIdx.z, h = blockIdx.y, qt = blockIdx.x;
  int wave = threadIdx.x >> 5;
  int lane = threadIdx.x & 31, lh = lane >> 4, lc = lane & 15;
  int bh = b * Hh + h;
  int qrow0 = qt * 64 + wave * 16;

  const _Float16* qcb = qc + ((size_t)bh * Qq + qrow0) * Kd;
  const _Float16* khb = kh + (size_t)bh * KL * Kd;
  const _Float16* vtb = vth + (size_t)bh * Kd * KL;
  const _Float16* bdb = bdraw + (size_t)bh * Qq * KL;
  const unsigned char* mkb = maskp + (size_t)b * Qq * KL;

  v16h aq0 = frag_ld(qcb, Kd);       // d 0..31
  v16h aq1 = frag_ld(qcb + 32, Kd);  // d 32..63
  v8f acc[4] = {};
  float mrow[8], lrow[8];
#pragma unroll
  for (int g = 0; g < 8; g++) { mrow[g] = -1e30f; lrow[g] = 0.f; }

  for (int j0 = 0; j0 < KL; j0 += 32) {
    v8f s0 = {}, s1 = {};
    s0 = wmma_f16(aq0, frag_ldb(khb + (size_t)j0 * Kd, Kd), s0);
    s0 = wmma_f16(aq1, frag_ldb(khb + (size_t)j0 * Kd + 32, Kd), s0);
    s1 = wmma_f16(aq0, frag_ldb(khb + (size_t)(j0 + 16) * Kd, Kd), s1);
    s1 = wmma_f16(aq1, frag_ldb(khb + (size_t)(j0 + 16) * Kd + 32, Kd), s1);

#pragma unroll
    for (int g = 0; g < 8; g++) {
      int i = qrow0 + g + lh * 8;
      int ja = j0 + lc, jb = j0 + 16 + lc;
      // Transformer-XL rel_shift: f = i*KL + j + Q; src = (f/(KL+1), f%(KL+1)-1)
      int fa = i * KL + ja + Qq, fb = i * KL + jb + Qq;
      int isa = fa / (KL + 1), jpa = fa % (KL + 1);
      int isb = fb / (KL + 1), jpb = fb % (KL + 1);
      float bda = jpa ? (float)bdb[(size_t)isa * KL + jpa - 1] : 0.f;
      float bdb_ = jpb ? (float)bdb[(size_t)isb * KL + jpb - 1] : 0.f;
      float v0 = (s0[g] + bda) * 0.125f;   // 1/sqrt(64)
      float v1 = (s1[g] + bdb_) * 0.125f;
      if (mkb[(size_t)i * KL + ja]) v0 = -1e9f;
      if (mkb[(size_t)i * KL + jb]) v1 = -1e9f;
      float mnew = fmaxf(mrow[g], rmax16(fmaxf(v0, v1)));
      float fac = expf(mrow[g] - mnew);
      float p0 = expf(v0 - mnew), p1 = expf(v1 - mnew);
      lrow[g] = lrow[g] * fac + rsum16(p0 + p1);
      mrow[g] = mnew;
#pragma unroll
      for (int ff = 0; ff < 4; ff++) acc[ff][g] *= fac;
      int rr = g + lh * 8;
      Pl[wave][rr][lc] = (_Float16)p0;
      Pl[wave][rr][16 + lc] = (_Float16)p1;
    }
    __syncthreads();
    v16h pa = frag_ld(&Pl[wave][0][0], 40);
#pragma unroll
    for (int ff = 0; ff < 4; ff++) {
      v16h bv = frag_ldb(vtb + (size_t)(ff * 16) * KL + j0, KL);
      acc[ff] = wmma_f16(pa, bv, acc[ff]);
    }
    __syncthreads();
  }

#pragma unroll
  for (int ff = 0; ff < 4; ff++) {
#pragma unroll
    for (int g = 0; g < 8; g++) {
      int i = qrow0 + g + lh * 8;
      int d = ff * 16 + lc;
      ctxh[((size_t)b * Qq + i) * HK + h * Kd + d] =
          (_Float16)(acc[ff][g] / lrow[g]);
    }
  }
}

// ------------------------------- LayerNorm ---------------------------------

__global__ __launch_bounds__(256) void ln_kernel(
    const float* __restrict__ x, const float* __restrict__ g,
    const float* __restrict__ bt, const float* __restrict__ npad,
    float* __restrict__ outF, _Float16* __restrict__ outH) {
  __shared__ float red[256];
  int row = blockIdx.x;
  const float* xr = x + (size_t)row * Dd;
  float s = 0.f;
  for (int i = threadIdx.x; i < Dd; i += 256) s += xr[i];
  red[threadIdx.x] = s;
  __syncthreads();
  for (int o = 128; o > 0; o >>= 1) {
    if (threadIdx.x < o) red[threadIdx.x] += red[threadIdx.x + o];
    __syncthreads();
  }
  float mu = red[0] / Dd;
  __syncthreads();
  float vs = 0.f;
  for (int i = threadIdx.x; i < Dd; i += 256) {
    float dte = xr[i] - mu;
    vs += dte * dte;
  }
  red[threadIdx.x] = vs;
  __syncthreads();
  for (int o = 128; o > 0; o >>= 1) {
    if (threadIdx.x < o) red[threadIdx.x] += red[threadIdx.x + o];
    __syncthreads();
  }
  float inv = rsqrtf(red[0] / Dd + 1e-5f);
  float np = npad[row];
  for (int i = threadIdx.x; i < Dd; i += 256) {
    float y = ((xr[i] - mu) * inv * g[i] + bt[i]) * np;
    outF[(size_t)row * Dd + i] = y;
    if (outH) outH[(size_t)row * Dd + i] = (_Float16)y;
  }
}

// ------------------------------- launcher ----------------------------------

extern "C" void kernel_launch(void* const* d_in, const int* in_sizes, int n_in,
                              void* d_out, int out_size, void* d_ws, size_t ws_size,
                              hipStream_t stream) {
  (void)in_sizes; (void)n_in; (void)out_size; (void)ws_size;
  const float* layer_input  = (const float*)d_in[0];
  const float* memory       = (const float*)d_in[1];
  const float* content_bias = (const float*)d_in[2];
  const float* position_bias= (const float*)d_in[3];
  const float* non_pad      = (const float*)d_in[4];
  const float* Wq = (const float*)d_in[5];
  const float* Wk = (const float*)d_in[6];
  const float* Wv = (const float*)d_in[7];
  const float* Wr = (const float*)d_in[8];
  const float* Wo = (const float*)d_in[9];
  const float* ln1_g = (const float*)d_in[10];
  const float* ln1_b = (const float*)d_in[11];
  const float* W1 = (const float*)d_in[12];
  const float* b1 = (const float*)d_in[13];
  const float* W2 = (const float*)d_in[14];
  const float* b2 = (const float*)d_in[15];
  const float* ln2_g = (const float*)d_in[16];
  const float* ln2_b = (const float*)d_in[17];
  const unsigned char* mask = (const unsigned char*)d_in[18];

  char* wsp = (char*)d_ws;
  auto alloc = [&](size_t elems, size_t esz) -> void* {
    void* p = (void*)wsp;
    wsp += (elems * esz + 255) & ~(size_t)255;
    return p;
  };
  _Float16* xh   = (_Float16*)alloc((size_t)Bb * Qq * Dd, 2);
  _Float16* kvh  = (_Float16*)alloc((size_t)Bb * KL * Dd, 2);
  _Float16* peh  = (_Float16*)alloc((size_t)KL * Dd, 2);
  _Float16* WqT  = (_Float16*)alloc((size_t)HK * Dd, 2);
  _Float16* WkT  = (_Float16*)alloc((size_t)HK * Dd, 2);
  _Float16* WvT  = (_Float16*)alloc((size_t)HK * Dd, 2);
  _Float16* WrT  = (_Float16*)alloc((size_t)HK * Dd, 2);
  _Float16* WoT  = (_Float16*)alloc((size_t)Dd * HK, 2);
  _Float16* W1T  = (_Float16*)alloc((size_t)Ff * Dd, 2);
  _Float16* W2T  = (_Float16*)alloc((size_t)Dd * Ff, 2);
  _Float16* qc   = (_Float16*)alloc((size_t)Bb * Hh * Qq * Kd, 2);
  _Float16* qp   = (_Float16*)alloc((size_t)Bb * Hh * Qq * Kd, 2);
  _Float16* kh   = (_Float16*)alloc((size_t)Bb * Hh * KL * Kd, 2);
  _Float16* vth  = (_Float16*)alloc((size_t)Bb * Hh * Kd * KL, 2);
  _Float16* rh   = (_Float16*)alloc((size_t)Hh * KL * Kd, 2);
  _Float16* bdraw= (_Float16*)alloc((size_t)Bb * Hh * Qq * KL, 2);
  _Float16* ctxh = (_Float16*)alloc((size_t)Bb * Qq * HK, 2);
  _Float16* out1h= (_Float16*)alloc((size_t)Bb * Qq * Dd, 2);
  _Float16* hbuf = (_Float16*)alloc((size_t)Bb * Qq * Ff, 2);
  float* pre1f   = (float*)alloc((size_t)Bb * Qq * Dd, 4);
  float* out1f   = (float*)alloc((size_t)Bb * Qq * Dd, 4);
  float* pre2f   = (float*)alloc((size_t)Bb * Qq * Dd, 4);

  auto blks = [](long n) { return (unsigned)((n + 255) / 256); };

  // prep: conversions, kv pack, positional encoding, weight transposes
  conv_f16<<<blks((long)Bb * Qq * Dd), 256, 0, stream>>>(layer_input, xh, (long)Bb * Qq * Dd);
  pack_kv<<<blks((long)Bb * KL * Dd), 256, 0, stream>>>(memory, layer_input, kvh);
  pe_kernel<<<blks((long)KL * Dd), 256, 0, stream>>>(peh);
  tw_kernel<<<blks((long)Dd * HK), 256, 0, stream>>>(Wq, WqT, Dd, HK);
  tw_kernel<<<blks((long)Dd * HK), 256, 0, stream>>>(Wk, WkT, Dd, HK);
  tw_kernel<<<blks((long)Dd * HK), 256, 0, stream>>>(Wv, WvT, Dd, HK);
  tw_kernel<<<blks((long)Dd * HK), 256, 0, stream>>>(Wr, WrT, Dd, HK);
  tw_kernel<<<blks((long)HK * Dd), 256, 0, stream>>>(Wo, WoT, HK, Dd);
  tw_kernel<<<blks((long)Dd * Ff), 256, 0, stream>>>(W1, W1T, Dd, Ff);
  tw_kernel<<<blks((long)Ff * Dd), 256, 0, stream>>>(W2, W2T, Ff, Dd);

  // projections
  gemm_wmma<1><<<dim3(HK / 64, (Bb * Qq) / 128, 1), 256, 0, stream>>>(
      xh, WqT, Dd, 0, 0, 1, 0, qc, qp, nullptr, content_bias, position_bias,
      nullptr, Qq, HK);
  gemm_wmma<2><<<dim3(HK / 64, (Bb * KL) / 128, 1), 256, 0, stream>>>(
      kvh, WkT, Dd, 0, 0, 1, 0, kh, nullptr, nullptr, nullptr, nullptr,
      nullptr, KL, HK);
  gemm_wmma<5><<<dim3(HK / 64, (Bb * KL) / 128, 1), 256, 0, stream>>>(
      kvh, WvT, Dd, 0, 0, 1, 0, vth, nullptr, nullptr, nullptr, nullptr,
      nullptr, KL, HK);
  gemm_wmma<2><<<dim3(HK / 64, KL / 128, 1), 256, 0, stream>>>(
      peh, WrT, Dd, 0, 0, 1, 0, rh, nullptr, nullptr, nullptr, nullptr,
      nullptr, KL, HK);

  // BD = (q + pos_bias) @ r^T, batched over (b,h)
  gemm_wmma<0><<<dim3(KL / 64, Qq / 128, Bb * Hh), 256, 0, stream>>>(
      qp, rh, Kd, (long)Qq * Kd, (long)KL * Kd, Hh, (long)Qq * KL, bdraw,
      nullptr, nullptr, nullptr, nullptr, nullptr, Qq, KL);

  // fused attention -> ctx (f16, [B,Q,H*Kd])
  attn_kernel<<<dim3(Qq / 64, Hh, Bb), 128, 0, stream>>>(qc, kh, vth, bdraw,
                                                         mask, ctxh);

  // out projection + residual, LN1
  gemm_wmma<4><<<dim3(Dd / 64, (Bb * Qq) / 128, 1), 256, 0, stream>>>(
      ctxh, WoT, HK, 0, 0, 1, 0, nullptr, nullptr, pre1f, nullptr, nullptr,
      layer_input, Qq, Dd);
  ln_kernel<<<Bb * Qq, 256, 0, stream>>>(pre1f, ln1_g, ln1_b, non_pad, out1f,
                                         out1h);

  // FFN
  gemm_wmma<3><<<dim3(Ff / 64, (Bb * Qq) / 128, 1), 256, 0, stream>>>(
      out1h, W1T, Dd, 0, 0, 1, 0, hbuf, nullptr, nullptr, b1, nullptr,
      nullptr, Qq, Ff);
  gemm_wmma<4><<<dim3(Dd / 64, (Bb * Qq) / 128, 1), 256, 0, stream>>>(
      hbuf, W2T, Ff, 0, 0, 1, 0, nullptr, nullptr, pre2f, b2, nullptr, out1f,
      Qq, Dd);
  ln_kernel<<<Bb * Qq, 256, 0, stream>>>(pre2f, ln2_g, ln2_b, non_pad,
                                         (float*)d_out, nullptr);
}
